// DeFUM_31001073943143
// MI455X (gfx1250) — matrix operand order repaired and checked by reading, not compiled, not run on hardware
//
#include <hip/hip_runtime.h>

// ---------------------------------------------------------------------------
// Types
// ---------------------------------------------------------------------------
typedef __attribute__((ext_vector_type(16))) __bf16 v16bf;
typedef __attribute__((ext_vector_type(8)))  float  v8f;
typedef __attribute__((ext_vector_type(4)))  unsigned int u32x4;
typedef __attribute__((ext_vector_type(4)))  unsigned int v4u;
typedef __attribute__((ext_vector_type(8)))  int v8i;
typedef __attribute__((ext_vector_type(4)))  int v4i;

#define DEV __device__ __forceinline__

#if defined(__has_builtin)
#  if __has_builtin(__builtin_amdgcn_tensor_load_to_lds)
#    define HAVE_TDM 1
#  endif
#endif

DEV __bf16 f2bf(float f) {
  unsigned u = __builtin_bit_cast(unsigned, f);
  unsigned r = (u + 0x7fffu + ((u >> 16) & 1u)) >> 16;
  return __builtin_bit_cast(__bf16, (unsigned short)r);
}

union FragU { v16bf v; u32x4 q[2]; __bf16 e[16]; };
union H8   { u32x4 q; unsigned short us[8]; };

// A fragment (16x32 bf16) from row-major [*, ld] at tile origin (row, k).
DEV v16bf frag_a(const __bf16* p, int ld, int row, int k, int lane) {
  const __bf16* b = p + (row + (lane & 15)) * ld + k + ((lane >> 4) << 3);
  FragU f;
  f.q[0] = *(const u32x4*)(b);
  f.q[1] = *(const u32x4*)(b + 16);
  return f.v;
}

// B fragment (32x16) for D=A*B with B = W^T, W stored row-major [N,K].
// Lane l holds output column n+(l&15); same K order as frag_a
// (lanes 0-15: K=k..k+7,k+16..k+23; lanes 16-31: K=k+8..k+15,k+24..k+31).
DEV v16bf frag_b_nt(const __bf16* p, int ld, int n, int k, int lane) {
  const __bf16* b = p + (n + (lane & 15)) * ld + k + ((lane >> 4) << 3);
  FragU f;
  f.q[0] = *(const u32x4*)(b);
  f.q[1] = *(const u32x4*)(b + 16);
  return f.v;
}

#define WMMA_BF16(a, b, c) \
  __builtin_amdgcn_wmma_f32_16x16x32_bf16(false, (a), false, (b), (short)0, (c), false, false)

// ---------------------------------------------------------------------------
// Tensor Data Mover: 2D bf16 tile load, global -> LDS.
// tile_d0 elems contiguous per line, tile_d1 lines at stride_elems,
// optional LDS padding of 4 dwords after every 16 dwords (64B data -> 80B row,
// i.e. exactly our padded stride-40 bf16 LDS layout).
// ---------------------------------------------------------------------------
#if defined(HAVE_TDM)
DEV void tdm_load_tile_bf16(const void* g, unsigned lds_off,
                            int tile_d0, int tile_d1,
                            long long stride_elems, int pad16) {
  unsigned long long ga = (unsigned long long)(size_t)g;
  v4u g0;
  g0[0] = 1u;                                              // count=1, user mode
  g0[1] = lds_off;                                         // lds_addr
  g0[2] = (unsigned)ga;                                    // global_addr[31:0]
  g0[3] = (unsigned)((ga >> 32) & 0x01ffffffu) | (2u << 30); // ga[56:32] | type=2
  unsigned w0 = 1u << 16;                                  // data_size = 2B
  if (pad16) w0 |= (1u << 20) | (3u << 22) | (3u << 25);   // pad 4 dw / 16 dw
  const unsigned td = 0x40000000u;                         // huge tensor dims
  const unsigned long long s0 = (unsigned long long)stride_elems;
  v8i g1;
  g1[0] = (int)w0;                                         // wg_mask=0, flags
  g1[1] = (int)((td & 0xffffu) << 16);                     // tensor_dim0 lo
  g1[2] = (int)((td >> 16) | ((td & 0xffffu) << 16));      // dim0 hi | dim1 lo
  g1[3] = (int)((td >> 16) | ((unsigned)tile_d0 << 16));   // dim1 hi | tile_dim0
  g1[4] = (int)((unsigned)tile_d1 & 0xffffu);              // tile_dim1, tile_dim2=0
  g1[5] = (int)(unsigned)(s0 & 0xffffffffu);               // dim0_stride lo
  g1[6] = (int)(unsigned)((s0 >> 32) & 0xffffu);           // dim0_stride hi
  g1[7] = 0;
  v4i z4 = {0, 0, 0, 0};
#  if __clang_major__ >= 23
  v8i z8 = {0, 0, 0, 0, 0, 0, 0, 0};
  __builtin_amdgcn_tensor_load_to_lds(g0, g1, z4, z4, z8, 0);
#  else
  __builtin_amdgcn_tensor_load_to_lds(g0, g1, z4, z4, 0);
#  endif
}
#endif

// ---------------------------------------------------------------------------
// Small helpers
// ---------------------------------------------------------------------------
__global__ void cvt_bf16_kernel(const float* __restrict__ s, __bf16* __restrict__ d, long long n) {
  long long i = (long long)blockIdx.x * blockDim.x + threadIdx.x;
  long long st = (long long)gridDim.x * blockDim.x;
  for (; i < n; i += st) d[i] = f2bf(s[i]);
}

__global__ void build_x_kernel(const float* __restrict__ a, const float* __restrict__ b,
                               float* __restrict__ xf, __bf16* __restrict__ xb) {
  long long i = (long long)blockIdx.x * 256 + threadIdx.x;      // 12582912 total
  long long d  = i % 768;
  long long sr = (i / 768) % 512;
  long long bb = i / (768LL * 512);
  float v = (sr < 256) ? a[bb * 196608 + sr * 768 + d]
                       : b[bb * 196608 + (sr - 256) * 768 + d];
  xf[i] = v;
  xb[i] = f2bf(v);
}

__global__ void build_logd_pad_kernel(const float* __restrict__ da, const float* __restrict__ db,
                                      const int* __restrict__ ma, const int* __restrict__ mb,
                                      float* __restrict__ logd, int* __restrict__ pad) {
  int i = blockIdx.x * 256 + threadIdx.x;                        // 16384 total
  int s = i & 511, b = i >> 9;
  float dv = (s < 256) ? da[b * 256 + s] : db[b * 256 + (s - 256)];
  int mv   = (s < 256) ? ma[b * 256 + s] : mb[b * 256 + (s - 256)];
  logd[i] = __logf(dv);
  pad[i]  = mv;
}

// ---------------------------------------------------------------------------
// GEMM  C = A @ W^T  (A[M,K] bf16 row-major, W[N,K] bf16 row-major)
// Block tile 128x128x32; 8 waves (4x2); wave tile 32x64 (2x4 WMMA frags).
// W tile staged via Tensor Data Mover (wave 0), A tile via b128 loads.
// EPI: 0 bias->bf16 | 1 bias+relu->bf16 | 2 bias->f32 | 4 defum scores->f32
// ---------------------------------------------------------------------------
template <int EPI>
__global__ __launch_bounds__(256)
void gemm_nt_kernel(const __bf16* __restrict__ A, int lda, long long sA,
                    const __bf16* __restrict__ W, int ldb, long long sB,
                    const float* __restrict__ bias,
                    float* __restrict__ Cf, __bf16* __restrict__ Cb,
                    int ldc, long long sC, int K,
                    const float* __restrict__ logd, const int* __restrict__ pad,
                    float scale, int Sdim) {
  __shared__ __bf16 As[128 * 40];
  __shared__ __bf16 Bs[128 * 40];
  const int tid = threadIdx.x, lane = tid & 31, w = tid >> 5;
  const int wm = w >> 1, wn = w & 1;
  const int m0 = blockIdx.y << 7, n0 = blockIdx.x << 7;
  const int z = blockIdx.z;
  A += (long long)z * sA;
  W += (long long)z * sB;

  v8f acc[2][4];
#pragma unroll
  for (int i = 0; i < 2; ++i)
#pragma unroll
    for (int j = 0; j < 4; ++j)
#pragma unroll
      for (int r = 0; r < 8; ++r) acc[i][j][r] = 0.f;

  const int r0 = tid >> 2, c0 = (tid & 3) << 3;   // A: 2 x 16B chunks per thread
#if defined(HAVE_TDM)
  const unsigned bs_off = (unsigned)(size_t)(const void*)&Bs[0];
#endif

  for (int k0 = 0; k0 < K; k0 += 32) {
    __syncthreads();
#if defined(HAVE_TDM)
    if (w == 0)
      tdm_load_tile_bf16(W + (long long)n0 * ldb + k0, bs_off, 32, 128, ldb, 1);
#else
    *(u32x4*)&Bs[r0 * 40 + c0]        = *(const u32x4*)(W + (long long)(n0 + r0) * ldb + k0 + c0);
    *(u32x4*)&Bs[(r0 + 64) * 40 + c0] = *(const u32x4*)(W + (long long)(n0 + r0 + 64) * ldb + k0 + c0);
#endif
    *(u32x4*)&As[r0 * 40 + c0]        = *(const u32x4*)(A + (long long)(m0 + r0) * lda + k0 + c0);
    *(u32x4*)&As[(r0 + 64) * 40 + c0] = *(const u32x4*)(A + (long long)(m0 + r0 + 64) * lda + k0 + c0);
    if (k0 + 32 < K)
      __builtin_prefetch((const void*)(A + (long long)(m0 + r0) * lda + k0 + 32 + c0), 0, 1);
#if defined(HAVE_TDM)
    if (w == 0) __builtin_amdgcn_s_wait_tensorcnt(0);
#endif
    __syncthreads();
    v16bf a0 = frag_a(As, 40, wm * 32, 0, lane);
    v16bf a1 = frag_a(As, 40, wm * 32 + 16, 0, lane);
#pragma unroll
    for (int j = 0; j < 4; ++j) {
      v16bf bb = frag_b_nt(Bs, 40, wn * 64 + j * 16, 0, lane);
      acc[0][j] = WMMA_BF16(a0, bb, acc[0][j]);
      acc[1][j] = WMMA_BF16(a1, bb, acc[1][j]);
    }
  }

  const long long cb = (long long)z * sC;
#pragma unroll
  for (int i = 0; i < 2; ++i) {
    const int rowb = m0 + wm * 32 + i * 16 + ((lane >> 4) << 3);
#pragma unroll
    for (int j = 0; j < 4; ++j) {
      const int col = n0 + wn * 64 + j * 16 + (lane & 15);
      float badd = 0.f, lj = 0.f, msk = 0.f;
      if (EPI == 0 || EPI == 1 || EPI == 2) badd = bias[col];
      if (EPI == 4) {
        lj  = logd[(long long)z * Sdim + col];
        msk = pad[(long long)z * Sdim + col] ? -1e9f : 0.f;
      }
#pragma unroll
      for (int r = 0; r < 8; ++r) {
        const int row = rowb + r;
        float v = acc[i][j][r];
        if (EPI == 4) {
          float li = logd[(long long)z * Sdim + row];
          v = v * scale + fabsf(li - lj) + msk;
        } else {
          v += badd;
        }
        if (EPI == 1) v = fmaxf(v, 0.f);
        const long long idx = cb + (long long)row * ldc + col;
        if (EPI == 0 || EPI == 1) Cb[idx] = f2bf(v);
        else                      Cf[idx] = v;
      }
    }
  }
}

// ---------------------------------------------------------------------------
// GEMM  C = A @ B  (A[M,K] bf16, B[K,N] bf16 natural), f32 out, no bias.
// B tile transposed in registers at staging time -> all frag loads are b128.
// ---------------------------------------------------------------------------
__global__ __launch_bounds__(256)
void gemm_nn_kernel(const __bf16* __restrict__ A, int lda, long long sA,
                    const __bf16* __restrict__ B, int ldb, long long sB,
                    float* __restrict__ Cf, int ldc, long long sC, int K) {
  __shared__ __bf16 As[128 * 40];
  __shared__ __bf16 Bt[128 * 40];   // transposed: Bt[n][k]
  const int tid = threadIdx.x, lane = tid & 31, w = tid >> 5;
  const int wm = w >> 1, wn = w & 1;
  const int m0 = blockIdx.y << 7, n0 = blockIdx.x << 7;
  const int z = blockIdx.z;
  A += (long long)z * sA;
  B += (long long)z * sB;

  v8f acc[2][4];
#pragma unroll
  for (int i = 0; i < 2; ++i)
#pragma unroll
    for (int j = 0; j < 4; ++j)
#pragma unroll
      for (int r = 0; r < 8; ++r) acc[i][j][r] = 0.f;

  const int r0 = tid >> 2, c0 = (tid & 3) << 3;
  const int kp = tid >> 4, nc = (tid & 15) << 3;   // k-pair x 8-wide n chunk
  for (int k0 = 0; k0 < K; k0 += 32) {
    __syncthreads();
    *(u32x4*)&As[r0 * 40 + c0]        = *(const u32x4*)(A + (long long)(m0 + r0) * lda + k0 + c0);
    *(u32x4*)&As[(r0 + 64) * 40 + c0] = *(const u32x4*)(A + (long long)(m0 + r0 + 64) * lda + k0 + c0);
    {
      H8 ra, rb;
      ra.q = *(const u32x4*)(B + (long long)(k0 + 2 * kp)     * ldb + n0 + nc);
      rb.q = *(const u32x4*)(B + (long long)(k0 + 2 * kp + 1) * ldb + n0 + nc);
#pragma unroll
      for (int i = 0; i < 8; ++i) {
        unsigned wv = (unsigned)ra.us[i] | ((unsigned)rb.us[i] << 16);
        *(unsigned*)&Bt[(nc + i) * 40 + 2 * kp] = wv;
      }
    }
    __syncthreads();
    v16bf a0 = frag_a(As, 40, wm * 32, 0, lane);
    v16bf a1 = frag_a(As, 40, wm * 32 + 16, 0, lane);
#pragma unroll
    for (int j = 0; j < 4; ++j) {
      v16bf bb = frag_b_nt(Bt, 40, wn * 64 + j * 16, 0, lane);
      acc[0][j] = WMMA_BF16(a0, bb, acc[0][j]);
      acc[1][j] = WMMA_BF16(a1, bb, acc[1][j]);
    }
  }

  const long long cb = (long long)z * sC;
#pragma unroll
  for (int i = 0; i < 2; ++i) {
    const int rowb = m0 + wm * 32 + i * 16 + ((lane >> 4) << 3);
#pragma unroll
    for (int j = 0; j < 4; ++j) {
      const int col = n0 + wn * 64 + j * 16 + (lane & 15);
#pragma unroll
      for (int r = 0; r < 8; ++r)
        Cf[cb + (long long)(rowb + r) * ldc + col] = acc[i][j][r];
    }
  }
}

// ---------------------------------------------------------------------------
// Fused multi-head flash attention over qkv [B,S,3*768] bf16, DH=96, S=512.
// Grid (S/64, H, B), 128 threads = 4 waves; wave owns 16 query rows.
// V staged transposed (Vt[d][j]) so P@V B-fragments are b128 loads.
// ---------------------------------------------------------------------------
__global__ __launch_bounds__(128)
void flash_mha_kernel(const __bf16* __restrict__ qkv, const int* __restrict__ pad,
                      __bf16* __restrict__ ctx, float scale) {
  const int b = blockIdx.z, h = blockIdx.y, qt = blockIdx.x;
  const int tid = threadIdx.x, lane = tid & 31, w = tid >> 5;
  __shared__ __bf16 Qs[64 * 104];
  __shared__ __bf16 Ks[64 * 104];
  __shared__ __bf16 Vt[96 * 72];    // transposed: Vt[d][j]
  __shared__ __bf16 Ps[4][16 * 72];

  const long long bbase = (long long)b * 512 * 2304;
  const __bf16* qp = qkv + bbase + (long long)(qt * 64) * 2304 + h * 96;
  for (int c = tid; c < 768; c += 128) {                   // 64 rows x 12 chunks
    int r = c / 12, cc = (c % 12) * 8;
    *(u32x4*)&Qs[r * 104 + cc] = *(const u32x4*)(qp + (long long)r * 2304 + cc);
  }
  const int* pad_b = pad + b * 512;

  float m_i[8], l_i[8];
  v8f oacc[6];
#pragma unroll
  for (int r = 0; r < 8; ++r) { m_i[r] = -3.0e38f; l_i[r] = 0.f; }
#pragma unroll
  for (int f = 0; f < 6; ++f)
#pragma unroll
    for (int r = 0; r < 8; ++r) oacc[f][r] = 0.f;

  for (int jt = 0; jt < 8; ++jt) {
    __syncthreads();
    const __bf16* kp = qkv + bbase + (long long)(jt * 64) * 2304 + 768 + h * 96;
    const __bf16* vp = kp + 768;
    for (int c = tid; c < 768; c += 128) {
      int r = c / 12, cc = (c % 12) * 8;
      *(u32x4*)&Ks[r * 104 + cc] = *(const u32x4*)(kp + (long long)r * 2304 + cc);
    }
    for (int c = tid; c < 384; c += 128) {                 // 32 j-pairs x 12 d-chunks
      int jp = c / 12, dc = (c % 12) * 8;
      H8 ra, rb;
      ra.q = *(const u32x4*)(vp + (long long)(2 * jp)     * 2304 + dc);
      rb.q = *(const u32x4*)(vp + (long long)(2 * jp + 1) * 2304 + dc);
#pragma unroll
      for (int i = 0; i < 8; ++i) {
        unsigned wv = (unsigned)ra.us[i] | ((unsigned)rb.us[i] << 16);
        *(unsigned*)&Vt[(dc + i) * 72 + 2 * jp] = wv;
      }
    }
    __syncthreads();

    // S = Q @ K^T  (16x64 per wave, contraction over DH=96)
    v8f sacc[4];
#pragma unroll
    for (int n = 0; n < 4; ++n)
#pragma unroll
      for (int r = 0; r < 8; ++r) sacc[n][r] = 0.f;
#pragma unroll
    for (int kk = 0; kk < 3; ++kk) {
      v16bf a = frag_a(Qs, 104, w * 16, kk * 32, lane);
#pragma unroll
      for (int n = 0; n < 4; ++n) {
        v16bf bb = frag_b_nt(Ks, 104, n * 16, kk * 32, lane);
        sacc[n] = WMMA_BF16(a, bb, sacc[n]);
      }
    }

    float mbias[4];
#pragma unroll
    for (int n = 0; n < 4; ++n) {
      int col = jt * 64 + n * 16 + (lane & 15);
      mbias[n] = pad_b[col] ? -1e9f : 0.f;
    }

    // online softmax (rows live across lanes of same half-wave)
    float mcur[8];
#pragma unroll
    for (int r = 0; r < 8; ++r) mcur[r] = -3.0e38f;
#pragma unroll
    for (int n = 0; n < 4; ++n)
#pragma unroll
      for (int r = 0; r < 8; ++r) {
        float v = sacc[n][r] * scale + mbias[n];
        sacc[n][r] = v;
        mcur[r] = fmaxf(mcur[r], v);
      }
#pragma unroll
    for (int r = 0; r < 8; ++r) {
      mcur[r] = fmaxf(mcur[r], __shfl_xor(mcur[r], 1, 32));
      mcur[r] = fmaxf(mcur[r], __shfl_xor(mcur[r], 2, 32));
      mcur[r] = fmaxf(mcur[r], __shfl_xor(mcur[r], 4, 32));
      mcur[r] = fmaxf(mcur[r], __shfl_xor(mcur[r], 8, 32));
    }
    float alpha[8], lsum[8];
#pragma unroll
    for (int r = 0; r < 8; ++r) {
      float mn = fmaxf(m_i[r], mcur[r]);
      alpha[r] = __expf(m_i[r] - mn);
      m_i[r] = mn;
      lsum[r] = 0.f;
    }
#pragma unroll
    for (int n = 0; n < 4; ++n)
#pragma unroll
      for (int r = 0; r < 8; ++r) {
        float p = __expf(sacc[n][r] - m_i[r]);
        sacc[n][r] = p;
        lsum[r] += p;
      }
#pragma unroll
    for (int r = 0; r < 8; ++r) {
      lsum[r] += __shfl_xor(lsum[r], 1, 32);
      lsum[r] += __shfl_xor(lsum[r], 2, 32);
      lsum[r] += __shfl_xor(lsum[r], 4, 32);
      lsum[r] += __shfl_xor(lsum[r], 8, 32);
      l_i[r] = l_i[r] * alpha[r] + lsum[r];
    }
#pragma unroll
    for (int f = 0; f < 6; ++f)
#pragma unroll
      for (int r = 0; r < 8; ++r) oacc[f][r] *= alpha[r];

    // P -> LDS (wave-private), re-enter as A operand for O += P @ V
    __bf16* pw = &Ps[w][0];
#pragma unroll
    for (int n = 0; n < 4; ++n)
#pragma unroll
      for (int r = 0; r < 8; ++r)
        pw[(((lane >> 4) << 3) + r) * 72 + n * 16 + (lane & 15)] = f2bf(sacc[n][r]);
#pragma unroll
    for (int kk = 0; kk < 2; ++kk) {
      v16bf a = frag_a(pw, 72, 0, kk * 32, lane);
#pragma unroll
      for (int f = 0; f < 6; ++f) {
        v16bf bb = frag_b_nt(Vt, 72, f * 16, kk * 32, lane);   // n=d, k=j
        oacc[f] = WMMA_BF16(a, bb, oacc[f]);
      }
    }
  }

#pragma unroll
  for (int r = 0; r < 8; ++r) {
    float inv = 1.f / l_i[r];
    int qrow = qt * 64 + w * 16 + ((lane >> 4) << 3) + r;
    long long base = ((long long)b * 512 + qrow) * 768 + h * 96 + (lane & 15);
#pragma unroll
    for (int f = 0; f < 6; ++f) ctx[base + f * 16] = f2bf(oacc[f][r] * inv);
  }
}

// ---------------------------------------------------------------------------
// Row softmax over 512 cols: f32 scores -> bf16 probabilities. Wave per row.
// ---------------------------------------------------------------------------
__global__ __launch_bounds__(256)
void softmax_kernel(const float* __restrict__ sc, __bf16* __restrict__ pr) {
  long long row = (long long)blockIdx.x * 8 + (threadIdx.x >> 5);
  int lane = threadIdx.x & 31;
  const float* p = sc + row * 512;
  float v[16];
  float mx = -3.0e38f;
#pragma unroll
  for (int i = 0; i < 16; ++i) { v[i] = p[i * 32 + lane]; mx = fmaxf(mx, v[i]); }
#pragma unroll
  for (int off = 16; off > 0; off >>= 1) mx = fmaxf(mx, __shfl_xor(mx, off, 32));
  float s = 0.f;
#pragma unroll
  for (int i = 0; i < 16; ++i) { v[i] = __expf(v[i] - mx); s += v[i]; }
#pragma unroll
  for (int off = 16; off > 0; off >>= 1) s += __shfl_xor(s, off, 32);
  float inv = 1.f / s;
#pragma unroll
  for (int i = 0; i < 16; ++i) pr[row * 512 + i * 32 + lane] = f2bf(v[i] * inv);
}

// ---------------------------------------------------------------------------
// out = LayerNorm(res + delta) over 768 cols. Wave per row; writes f32+bf16.
// ---------------------------------------------------------------------------
__global__ __launch_bounds__(256)
void add_ln_kernel(const float* __restrict__ res, const float* __restrict__ delta,
                   const float* __restrict__ g, const float* __restrict__ be,
                   float* __restrict__ outF, __bf16* __restrict__ outB) {
  long long row = (long long)blockIdx.x * 8 + (threadIdx.x >> 5);
  int lane = threadIdx.x & 31;
  const float* rr = res + row * 768;
  const float* dd = delta + row * 768;
  float v[24], s = 0.f, s2 = 0.f;
#pragma unroll
  for (int i = 0; i < 24; ++i) {
    int c = i * 32 + lane;
    float x = rr[c] + dd[c];
    v[i] = x; s += x; s2 += x * x;
  }
#pragma unroll
  for (int off = 16; off > 0; off >>= 1) {
    s  += __shfl_xor(s, off, 32);
    s2 += __shfl_xor(s2, off, 32);
  }
  float mean = s * (1.f / 768.f);
  float var  = s2 * (1.f / 768.f) - mean * mean;
  float rs   = rsqrtf(var + 1e-5f);
#pragma unroll
  for (int i = 0; i < 24; ++i) {
    int c = i * 32 + lane;
    float y = (v[i] - mean) * rs * g[c] + be[c];
    outF[row * 768 + c] = y;
    outB[row * 768 + c] = f2bf(y);
  }
}

// ---------------------------------------------------------------------------
// Host orchestration
// ---------------------------------------------------------------------------
extern "C" void kernel_launch(void* const* d_in, const int* in_sizes, int n_in,
                              void* d_out, int out_size, void* d_ws, size_t ws_size,
                              hipStream_t stream) {
  (void)in_sizes; (void)n_in; (void)out_size; (void)ws_size;
  constexpr int Bb = 32, S = 512, D = 768, DFF = 2048, L = 4, D3 = 3 * 768;
  constexpr long long MROWS = (long long)Bb * S;             // 16384
  const float inv_sqrt_d  = 0.03608439182435161f;            // 1/sqrt(768)
  const float inv_sqrt_dh = 0.10206207261596575f;            // 1/sqrt(96)

  const float* ocr   = (const float*)d_in[0];
  const float* obj   = (const float*)d_in[1];
  const float* docr  = (const float*)d_in[2];
  const float* dobj  = (const float*)d_in[3];
  const int*   mocr  = (const int*)d_in[4];
  const int*   mobj  = (const int*)d_in[5];
  const float* dq_w  = (const float*)d_in[6];
  const float* dq_b  = (const float*)d_in[7];
  const float* dk_w  = (const float*)d_in[8];
  const float* dk_b  = (const float*)d_in[9];
  const float* dv_w  = (const float*)d_in[10];
  const float* dv_b  = (const float*)d_in[11];
  const float* ln0_g = (const float*)d_in[12];
  const float* ln0_b = (const float*)d_in[13];
  const float* ain_w = (const float*)d_in[14];
  const float* ain_b = (const float*)d_in[15];
  const float* aout_w = (const float*)d_in[16];
  const float* aout_b = (const float*)d_in[17];
  const float* ln1_g = (const float*)d_in[18];
  const float* ln1_b = (const float*)d_in[19];
  const float* ff1_w = (const float*)d_in[20];
  const float* ff1_b = (const float*)d_in[21];
  const float* ff2_w = (const float*)d_in[22];
  const float* ff2_b = (const float*)d_in[23];
  const float* ln2_g = (const float*)d_in[24];
  const float* ln2_b = (const float*)d_in[25];

  // workspace carve-up (256B aligned slabs)
  char* ws = (char*)d_ws;
  size_t off = 0;
  auto take = [&](size_t bytes) { size_t o = off; off += (bytes + 255) & ~(size_t)255; return o; };
  const long long ND = MROWS * D;                            // 12582912
  float*  XF   = (float*)(ws + take(ND * 4));
  float*  HF   = (float*)(ws + take(ND * 4));
  float*  OF   = (float*)(ws + take(ND * 4));
  __bf16* XB   = (__bf16*)(ws + take(ND * 2));
  __bf16* QKVB = (__bf16*)(ws + take(MROWS * (long long)D3 * 2));
  __bf16* CTXB = (__bf16*)(ws + take(ND * 2));
  size_t  oFFB = take((size_t)MROWS * DFF * 2);              // aliases scores+probs
  __bf16* FFB  = (__bf16*)(ws + oFFB);
  float*  SC   = (float*)(ws + oFFB);                                   // [B,S,S] f32
  __bf16* PR   = (__bf16*)(ws + oFFB + (size_t)Bb * S * S * 4);         // [B,S,S] bf16
  float*  LOGD = (float*)(ws + take((size_t)Bb * S * 4));
  int*    PAD  = (int*)(ws + take((size_t)Bb * S * 4));
  __bf16* WDQ  = (__bf16*)(ws + take((size_t)D * D * 2));
  __bf16* WDK  = (__bf16*)(ws + take((size_t)D * D * 2));
  __bf16* WDV  = (__bf16*)(ws + take((size_t)D * D * 2));
  __bf16* WIN  = (__bf16*)(ws + take((size_t)L * D3 * D * 2));
  __bf16* WOUT = (__bf16*)(ws + take((size_t)L * D * D * 2));
  __bf16* WFF1 = (__bf16*)(ws + take((size_t)L * DFF * D * 2));
  __bf16* WFF2 = (__bf16*)(ws + take((size_t)L * D * DFF * 2));

  // --- weight conversion to bf16 ---
  cvt_bf16_kernel<<<1024, 256, 0, stream>>>(dq_w, WDQ, (long long)D * D);
  cvt_bf16_kernel<<<1024, 256, 0, stream>>>(dk_w, WDK, (long long)D * D);
  cvt_bf16_kernel<<<1024, 256, 0, stream>>>(dv_w, WDV, (long long)D * D);
  cvt_bf16_kernel<<<2048, 256, 0, stream>>>(ain_w, WIN, (long long)L * D3 * D);
  cvt_bf16_kernel<<<2048, 256, 0, stream>>>(aout_w, WOUT, (long long)L * D * D);
  cvt_bf16_kernel<<<2048, 256, 0, stream>>>(ff1_w, WFF1, (long long)L * DFF * D);
  cvt_bf16_kernel<<<2048, 256, 0, stream>>>(ff2_w, WFF2, (long long)L * D * DFF);

  // --- input assembly ---
  build_x_kernel<<<49152, 256, 0, stream>>>(ocr, obj, XF, XB);
  build_logd_pad_kernel<<<64, 256, 0, stream>>>(docr, dobj, mocr, mobj, LOGD, PAD);

  // --- DeFum single-head depth-biased attention ---
  gemm_nt_kernel<0><<<dim3(6, 128, 1), 256, 0, stream>>>(
      XB, D, 0, WDQ, D, 0, dq_b, nullptr, QKVB + 0, D3, 0, D, nullptr, nullptr, 0.f, 0);
  gemm_nt_kernel<0><<<dim3(6, 128, 1), 256, 0, stream>>>(
      XB, D, 0, WDK, D, 0, dk_b, nullptr, QKVB + 768, D3, 0, D, nullptr, nullptr, 0.f, 0);
  gemm_nt_kernel<0><<<dim3(6, 128, 1), 256, 0, stream>>>(
      XB, D, 0, WDV, D, 0, dv_b, nullptr, QKVB + 1536, D3, 0, D, nullptr, nullptr, 0.f, 0);
  // scores = q k^T / sqrt(D) + |log d_i - log d_j| + mask   [B,S,S] f32
  gemm_nt_kernel<4><<<dim3(4, 4, Bb), 256, 0, stream>>>(
      QKVB + 0, D3, (long long)S * D3, QKVB + 768, D3, (long long)S * D3,
      nullptr, SC, nullptr, S, (long long)S * S, D, LOGD, PAD, inv_sqrt_d, S);
  softmax_kernel<<<2048, 256, 0, stream>>>(SC, PR);
  // attn = P @ v  -> OF f32
  gemm_nn_kernel<<<dim3(6, 4, Bb), 256, 0, stream>>>(
      PR, S, (long long)S * S, QKVB + 1536, D3, (long long)S * D3,
      OF, D, (long long)S * D, S);
  // h = LN(x + attn)
  add_ln_kernel<<<2048, 256, 0, stream>>>(XF, OF, ln0_g, ln0_b, HF, XB);

  // --- transformer encoder layers ---
  for (int i = 0; i < L; ++i) {
    gemm_nt_kernel<0><<<dim3(18, 128, 1), 256, 0, stream>>>(
        XB, D, 0, WIN + (size_t)i * D3 * D, D, 0, ain_b + (size_t)i * D3,
        nullptr, QKVB, D3, 0, D, nullptr, nullptr, 0.f, 0);
    flash_mha_kernel<<<dim3(8, 8, Bb), 128, 0, stream>>>(QKVB, PAD, CTXB, inv_sqrt_dh);
    gemm_nt_kernel<2><<<dim3(6, 128, 1), 256, 0, stream>>>(
        CTXB, D, 0, WOUT + (size_t)i * D * D, D, 0, aout_b + (size_t)i * D,
        OF, nullptr, D, 0, D, nullptr, nullptr, 0.f, 0);
    add_ln_kernel<<<2048, 256, 0, stream>>>(HF, OF, ln1_g + (size_t)i * D, ln1_b + (size_t)i * D, HF, XB);
    gemm_nt_kernel<1><<<dim3(16, 128, 1), 256, 0, stream>>>(
        XB, D, 0, WFF1 + (size_t)i * DFF * D, D, 0, ff1_b + (size_t)i * DFF,
        nullptr, FFB, DFF, 0, D, nullptr, nullptr, 0.f, 0);
    gemm_nt_kernel<2><<<dim3(6, 128, 1), 256, 0, stream>>>(
        FFB, DFF, 0, WFF2 + (size_t)i * D * DFF, DFF, 0, ff2_b + (size_t)i * D,
        OF, nullptr, D, 0, DFF, nullptr, nullptr, 0.f, 0);
    float* lnOut = (i == L - 1) ? (float*)d_out : HF;
    add_ln_kernel<<<2048, 256, 0, stream>>>(HF, OF, ln2_g + (size_t)i * D, ln2_b + (size_t)i * D, lnOut, XB);
  }
}
